// VQINR_4363686772945
// MI455X (gfx1250) — matrix operand order repaired and checked by reading, not compiled, not run on hardware
//
#include <hip/hip_runtime.h>
#include <hip/hip_bf16.h>

// ---------- constants ----------
#define BATCH 16
#define NPTS  4096
#define LD    256
#define NC    1024
#define HID   256
#define NL    5
#define NV    4
#define OMEGA 30.0f
#define CC    0.25f

typedef __attribute__((ext_vector_type(16))) _Float16 v16h;
typedef __attribute__((ext_vector_type(8)))  float    v8f;
typedef __attribute__((ext_vector_type(4)))  float    v4f;
typedef __attribute__((ext_vector_type(4)))  unsigned int v4u;
typedef __attribute__((ext_vector_type(4)))  int      v4i;

union Frag16 { v4u u[2]; v16h h; };

#define AS1 __attribute__((address_space(1)))
#define AS3 __attribute__((address_space(3)))

#if defined(__AMDGCN__) && __has_builtin(__builtin_amdgcn_global_load_async_to_lds_b128) && __has_builtin(__builtin_amdgcn_s_wait_asynccnt)
#define HAVE_ASYNC_LDS 1
#else
#define HAVE_ASYNC_LDS 0
#endif

// =====================================================================
// Kernel 1: transpose Wh [4][256][256] f32 -> Wt [4][n][k] f16 (B-operand
// is loaded here as 16 contiguous K per lane/column).
// =====================================================================
__global__ void wt_transpose_kernel(const float* __restrict__ Wh,
                                    _Float16* __restrict__ Wt) {
    int l = blockIdx.x >> 8;        // 0..3
    int k = blockIdx.x & 255;       // 0..255
    int n = threadIdx.x;            // 0..255 (coalesced read)
    Wt[((size_t)(l * 256 + n)) * 256 + k] =
        (_Float16)Wh[((size_t)(l * 256 + k)) * 256 + n];
}

// =====================================================================
// Kernel 2: residual VQ. One block, 512 threads = 16 waves.
// Distances R[16,256] x E^T[256,1024] via v_wmma_f32_16x16x32_f16.
// Outputs: zsum_ws [16][256] f32, vq_loss -> out[196608].
// =====================================================================
#define RST 260  // padded residual row stride (floats)

__global__ __launch_bounds__(512) void vq_kernel(
    const int*   __restrict__ lidx,     // [16]
    const float* __restrict__ latents,  // [NI][NV][LD]
    const float* __restrict__ emb,      // [NV][NC][LD]
    float* __restrict__ zsum_ws,        // [16][256]
    float* __restrict__ out)            // d_out (loss at [196608])
{
    extern __shared__ float smemf[];
    float* Rs    = smemf;               // 16*260 = 4160
    float* Zsum  = smemf + 4160;        // 16*256 = 4096
    float* Dots  = smemf + 8256;        // 16*1024 = 16384
    float* e2    = smemf + 24640;       // 1024
    float* lred  = smemf + 25664;       // 512
    float* lossT = smemf + 26176;       // 1
    int*   idxA  = (int*)(smemf + 26177); // 16

    const int t = threadIdx.x;
    for (int i = t; i < BATCH * LD; i += 512) {
        int b = i >> 8, d = i & 255;
        int img = lidx[b];
        float s = 0.f;
        #pragma unroll
        for (int v = 0; v < NV; ++v)
            s += latents[((size_t)(img * NV + v)) * LD + d];
        Rs[b * RST + d] = s;
        Zsum[i] = 0.f;
    }
    if (t == 0) *lossT = 0.f;
    __syncthreads();

    const int wave = t >> 5, lane = t & 31;
    const int hi = lane >> 4, lm = lane & 15;

    for (int s = 0; s < NV; ++s) {
        const float* E = emb + (size_t)s * NC * LD;
        for (int c = t; c < NC; c += 512) {
            float acc = 0.f;
            const float* ep = E + (size_t)c * LD;
            for (int k = 0; k < LD; ++k) { float e = ep[k]; acc += e * e; }
            e2[c] = acc;
        }
        __syncthreads();

        const int codeBase = wave * 64;
        v8f zero = {0.f,0.f,0.f,0.f,0.f,0.f,0.f,0.f};
        v8f acc[4];
        #pragma unroll
        for (int nt = 0; nt < 4; ++nt) acc[nt] = zero;

        #pragma unroll
        for (int ks = 0; ks < 8; ++ks) {
            const int kBase = ks * 32;
            v4f a0 = *(const v4f*)&Rs[lm * RST + kBase + hi * 16];
            v4f a1 = *(const v4f*)&Rs[lm * RST + kBase + hi * 16 + 4];
            v4f a2 = *(const v4f*)&Rs[lm * RST + kBase + hi * 16 + 8];
            v4f a3 = *(const v4f*)&Rs[lm * RST + kBase + hi * 16 + 12];
            v16h a;
            #pragma unroll
            for (int i = 0; i < 4; ++i) {
                a[i]      = (_Float16)a0[i];
                a[4 + i]  = (_Float16)a1[i];
                a[8 + i]  = (_Float16)a2[i];
                a[12 + i] = (_Float16)a3[i];
            }
            #pragma unroll
            for (int nt = 0; nt < 4; ++nt) {
                int n  = codeBase + nt * 16 + lm;
                int kB = kBase + hi * 16;
                const float* ep = E + (size_t)n * LD + kB;
                v16h bf;
                #pragma unroll
                for (int q = 0; q < 4; ++q) {
                    v4f e4 = *(const v4f*)(ep + q * 4);
                    bf[q * 4 + 0] = (_Float16)e4[0];
                    bf[q * 4 + 1] = (_Float16)e4[1];
                    bf[q * 4 + 2] = (_Float16)e4[2];
                    bf[q * 4 + 3] = (_Float16)e4[3];
                }
                acc[nt] = __builtin_amdgcn_wmma_f32_16x16x32_f16(
                    false, a, false, bf, (short)0, acc[nt], false, false);
            }
        }
        #pragma unroll
        for (int nt = 0; nt < 4; ++nt) {
            int code = codeBase + nt * 16 + lm;
            float ee = e2[code];
            #pragma unroll
            for (int r = 0; r < 8; ++r) {
                int bb = r + 8 * hi;   // C layout: M = r + 8*(lane>=16)
                Dots[bb * NC + code] = ee - 2.0f * acc[nt][r];
            }
        }
        __syncthreads();

        {   // argmin: wave w handles batch row w
            int b = wave;
            float best = 3.0e38f; int bi = 0;
            for (int c = lane; c < NC; c += 32) {
                float v = Dots[b * NC + c];
                if (v < best) { best = v; bi = c; }
            }
            #pragma unroll
            for (int off = 16; off >= 1; off >>= 1) {
                float ov = __shfl_down(best, off);
                int   oi = __shfl_down(bi, off);
                if (ov < best || (ov == best && oi < bi)) { best = ov; bi = oi; }
            }
            if (lane == 0) idxA[b] = bi;
        }
        __syncthreads();

        float part = 0.f;
        for (int i = t; i < BATCH * LD; i += 512) {
            int b = i >> 8, d = i & 255;
            float q = E[(size_t)idxA[b] * LD + d];
            float r = Rs[b * RST + d];
            float diff = q - r;
            part += diff * diff;
            Zsum[i] += q;
            Rs[b * RST + d] = r - q;
        }
        lred[t] = part;
        __syncthreads();
        for (int off = 256; off >= 1; off >>= 1) {
            if (t < off) lred[t] += lred[t + off];
            __syncthreads();
        }
        if (t == 0) *lossT += CC * (lred[0] / (float)(BATCH * LD));
        __syncthreads();
    }

    for (int i = t; i < BATCH * LD; i += 512) zsum_ws[i] = Zsum[i];
    if (t == 0) out[(size_t)BATCH * NPTS * 3] = *lossT / (float)NV;
}

// =====================================================================
// Kernel 3: FiLM mods[b][l][h] = sum_d z[b][d]*mod_W[l][d][h] + mod_b[l][h]
// =====================================================================
__global__ __launch_bounds__(256) void mods_kernel(
    const float* __restrict__ zsum,   // [16][256]
    const float* __restrict__ mod_W,  // [5][256][512]
    const float* __restrict__ mod_b,  // [5][512]
    float* __restrict__ mods)         // [16][5][512]
{
    __shared__ float Zs[BATCH * LD];
    int t = threadIdx.x;
    int l = blockIdx.x >> 1;
    int h = (blockIdx.x & 1) * 256 + t;
    for (int i = t; i < BATCH * LD; i += 256) Zs[i] = zsum[i];
    __syncthreads();
    float acc[BATCH];
    #pragma unroll
    for (int b = 0; b < BATCH; ++b) acc[b] = 0.f;
    for (int k = 0; k < LD; ++k) {
        float w = mod_W[((size_t)(l * LD + k)) * 512 + h];
        #pragma unroll
        for (int b = 0; b < BATCH; ++b) acc[b] += Zs[b * LD + k] * w;
    }
    float mb = mod_b[l * 512 + h];
    #pragma unroll
    for (int b = 0; b < BATCH; ++b)
        mods[((size_t)(b * NL + l)) * 512 + h] = acc[b] + mb;
}

// =====================================================================
// Kernel 4: SIREN decoder. 512 blocks (b x 128-row tile), 512 threads
// = 16 waves; wave (r,c) owns 16 rows x 128 cols = 8 WMMA tiles.
// Layer weights staged to LDS (async-to-LDS) once per block/layer.
// =====================================================================
#define XST 264  // LDS row stride in halfs: 528B => 4-bank skew per row
#define ROWS 128

__global__ __launch_bounds__(512) void siren_kernel(
    const float*    __restrict__ coords,  // [16][4096][2]
    const float*    __restrict__ W0,      // [2][256]
    const float*    __restrict__ b0,      // [256]
    const float*    __restrict__ bh,      // [4][256]
    const float*    __restrict__ Wl,      // [256][3]
    const float*    __restrict__ bl,      // [3]
    const float*    __restrict__ mods,    // [16][5][512]
    const _Float16* __restrict__ Wt,      // [4][n=256][k=256] f16
    float* __restrict__ out)              // [16][4096][3]
{
    extern __shared__ char smem[];
    _Float16 (*Xs)[XST]   = (_Float16(*)[XST])smem;                       // 128 x 264
    _Float16 (*Wlds)[XST] = (_Float16(*)[XST])(smem + ROWS * XST * 2);    // 256 x 264

    const int t  = threadIdx.x;
    const int b  = blockIdx.x >> 5;          // 512 blocks = 16 images x 32 tiles
    const int p0 = (blockIdx.x & 31) * ROWS;

    // ---- layer 0: x = sin(30*(coords@W0 + b0)) ----
    {
        int col = t & 255, rh = t >> 8;
        float w0a = W0[col], w0b = W0[256 + col], bb = b0[col];
        for (int r = 0; r < 64; ++r) {
            int row = rh * 64 + r;
            const float* cp = coords + ((size_t)(b * NPTS + p0 + row)) * 2;
            float pre = cp[0] * w0a + cp[1] * w0b + bb;
            Xs[row][col] = (_Float16)__sinf(OMEGA * pre);
        }
    }
    __syncthreads();

    const int wave = t >> 5, lane = t & 31;
    const int hi = lane >> 4, lm = lane & 15;
    const int rowBase = (wave >> 1) * 16;    // 8 row stripes
    const int colBase = (wave & 1) * 128;    // 2 col stripes

    for (int li = 1; li < NL; ++li) {
        const _Float16* Wlayer = Wt + (size_t)(li - 1) * 256 * 256;

        // ---- stage layer weights into LDS (256 rows x 256 halfs, padded) ----
        #pragma unroll
        for (int it = 0; it < 16; ++it) {
            int chunk = it * 512 + t;        // 8192 x b128 chunks
            int n = chunk >> 5, kc = (chunk & 31) * 8;
#if HAVE_ASYNC_LDS
            __builtin_amdgcn_global_load_async_to_lds_b128(
                (AS1 v4i*)(Wlayer + (size_t)n * 256 + kc),
                (AS3 v4i*)(&Wlds[n][kc]), 0, 0);
#else
            *(v4u*)&Wlds[n][kc] = *(const v4u*)(Wlayer + (size_t)n * 256 + kc);
#endif
        }
#if HAVE_ASYNC_LDS
        __builtin_amdgcn_s_wait_asynccnt(0);
#endif
        __syncthreads();

        // ---- GEMM: 8 k-steps x 8 n-tiles per wave ----
        v8f zero = {0.f,0.f,0.f,0.f,0.f,0.f,0.f,0.f};
        v8f acc[8];
        #pragma unroll
        for (int nt = 0; nt < 8; ++nt) acc[nt] = zero;

        #pragma unroll
        for (int ks = 0; ks < 8; ++ks) {
            const int kBase = ks * 32;
            Frag16 a;   // A: lanes 0-15 row lm K{0..15}; lanes 16-31 K{16..31}
            a.u[0] = *(const v4u*)&Xs[rowBase + lm][kBase + hi * 16];
            a.u[1] = *(const v4u*)&Xs[rowBase + lm][kBase + hi * 16 + 8];
            #pragma unroll
            for (int nt = 0; nt < 8; ++nt) {
                int n = colBase + nt * 16 + lm;
                Frag16 bf;  // B: 16 contiguous K per lane for column n
                bf.u[0] = *(const v4u*)&Wlds[n][kBase + hi * 16];
                bf.u[1] = *(const v4u*)&Wlds[n][kBase + hi * 16 + 8];
                acc[nt] = __builtin_amdgcn_wmma_f32_16x16x32_f16(
                    false, a.h, false, bf.h, (short)0, acc[nt], false, false);
            }
        }
        __syncthreads();  // all Xs/Wlds reads done before overwrites
        #pragma unroll
        for (int nt = 0; nt < 8; ++nt) {
            int c = colBase + nt * 16 + lm;
            float g  = 1.0f + mods[((size_t)(b * NL + li)) * 512 + c];
            float be = mods[((size_t)(b * NL + li)) * 512 + 256 + c];
            float bias = bh[(li - 1) * 256 + c];
            #pragma unroll
            for (int r = 0; r < 8; ++r) {
                float pre = (acc[nt][r] + bias) * g + be;
                Xs[rowBase + r + 8 * hi][c] = (_Float16)__sinf(OMEGA * pre);
            }
        }
        __syncthreads();
    }

    // ---- final layer: out = x @ Wl + bl ----
    {
        int row = t >> 2, c = t & 3;
        if (c < 3) {
            float s = 0.f;
            for (int k = 0; k < HID; ++k)
                s += (float)Xs[row][k] * Wl[k * 3 + c];
            out[((size_t)(b * NPTS + p0 + row)) * 3 + c] = s + bl[c];
        }
    }
}

// =====================================================================
// launch
// =====================================================================
extern "C" void kernel_launch(void* const* d_in, const int* in_sizes, int n_in,
                              void* d_out, int out_size, void* d_ws, size_t ws_size,
                              hipStream_t stream) {
    const float* coords  = (const float*)d_in[0];
    const int*   lidx    = (const int*)  d_in[1];
    const float* latents = (const float*)d_in[2];
    const float* emb     = (const float*)d_in[3];
    const float* mod_W   = (const float*)d_in[4];
    const float* mod_b   = (const float*)d_in[5];
    const float* W0      = (const float*)d_in[6];
    const float* b0      = (const float*)d_in[7];
    const float* Wh      = (const float*)d_in[8];
    const float* bh      = (const float*)d_in[9];
    const float* Wl      = (const float*)d_in[10];
    const float* bl      = (const float*)d_in[11];
    float* out = (float*)d_out;

    float*    zsum_ws = (float*)d_ws;                            // 16*256
    float*    mods_ws = zsum_ws + BATCH * LD;                    // 16*5*512
    _Float16* wt_ws   = (_Float16*)(mods_ws + BATCH * NL * 512); // 4*256*256 f16

    size_t vqSmem    = (size_t)(26177 + 16) * sizeof(float);          // ~105KB
    size_t sirenSmem = (size_t)(ROWS + 256) * XST * sizeof(_Float16); // ~198KB

    // allow >64KB dynamic LDS (host-side attribute set; not a stream op)
    (void)hipFuncSetAttribute((const void*)vq_kernel,
        hipFuncAttributeMaxDynamicSharedMemorySize, (int)vqSmem);
    (void)hipFuncSetAttribute((const void*)siren_kernel,
        hipFuncAttributeMaxDynamicSharedMemorySize, (int)sirenSmem);

    wt_transpose_kernel<<<dim3(1024), dim3(256), 0, stream>>>(Wh, wt_ws);

    vq_kernel<<<dim3(1), dim3(512), vqSmem, stream>>>(lidx, latents, emb, zsum_ws, out);

    mods_kernel<<<dim3(NL * 2), dim3(256), 0, stream>>>(zsum_ws, mod_W, mod_b, mods_ws);

    siren_kernel<<<dim3(BATCH * (NPTS / ROWS)), dim3(512), sirenSmem, stream>>>(
        coords, W0, b0, bh, Wl, bl, mods_ws, wt_ws, out);
}
